// CrystalDiffusionModel_83571473646096
// MI455X (gfx1250) — matrix operand order repaired and checked by reading, not compiled, not run on hardware
//
#include <hip/hip_runtime.h>
#include <hip/hip_bf16.h>
#include <math.h>

#define NN 10000
#define EE 160000
#define BBATCH 16
#define HH 256
#define TDD 128
#define NBLK 4
#define NLAY 3
#define NAT_ 100

typedef _Float16 v16h __attribute__((ext_vector_type(16)));
typedef _Float16 h8   __attribute__((ext_vector_type(8)));
typedef float    v8f  __attribute__((ext_vector_type(8)));

__device__ __forceinline__ float silu_f(float x) { return x / (1.0f + expf(-x)); }

__device__ __forceinline__ v16h cat16(h8 lo, h8 hi) {
  return __builtin_shufflevector(lo, hi, 0,1,2,3,4,5,6,7,8,9,10,11,12,13,14,15);
}

__device__ __forceinline__ v8f wmma_f16(v16h a, v16h b, v8f c) {
  return __builtin_amdgcn_wmma_f32_16x16x32_f16(false, a, false, b, (short)0, c, false, false);
}

// ---------------------------------------------------------------- small utils
__global__ void zero_f32_kernel(float* __restrict__ p, int n) {
  int i = blockIdx.x * 256 + threadIdx.x;
  if (i < n) p[i] = 0.0f;
}

__global__ void embed_kernel(const int* __restrict__ xa, const float* __restrict__ emb,
                             float* __restrict__ h, _Float16* __restrict__ hh) {
  int i = blockIdx.x * 256 + threadIdx.x;       // i < NN*HH exactly
  int n = i >> 8, j = i & 255;
  float v = emb[xa[n] * HH + j];
  h[i] = v; hh[i] = (_Float16)v;
}

__global__ void dist_kernel(const float* __restrict__ pos, const int* __restrict__ eidx,
                            float* __restrict__ distv) {
  int e = blockIdx.x * 256 + threadIdx.x;
  if (e < EE) {
    int r = eidx[e], c = eidx[EE + e];
    float dx = pos[r*3+0] - pos[c*3+0];
    float dy = pos[r*3+1] - pos[c*3+1];
    float dz = pos[r*3+2] - pos[c*3+2];
    distv[e] = sqrtf(dx*dx + dy*dy + dz*dz);
  }
}

__global__ void count_kernel(const int* __restrict__ eidx, float* __restrict__ cnt) {
  int e = blockIdx.x * 256 + threadIdx.x;
  if (e < EE) atomicAdd(&cnt[eidx[EE + e]], 1.0f);
}

__global__ void invcnt_kernel(float* __restrict__ cnt, int n) {
  int i = blockIdx.x * 256 + threadIdx.x;
  if (i < n) cnt[i] = 1.0f / fmaxf(cnt[i], 1.0f);
}

__global__ void addtp_kernel(float* __restrict__ h, _Float16* __restrict__ hh,
                             const float* __restrict__ tp, const int* __restrict__ bidx) {
  int i = blockIdx.x * 256 + threadIdx.x;       // i < NN*HH exactly
  int n = i >> 8, j = i & 255;
  float v = h[i] + tp[bidx[n] * HH + j];
  h[i] = v; hh[i] = (_Float16)v;
}

// out[n*K + k] = (f16) in[k*Nc + n]   (weight transpose + convert)
__global__ void transpose_kernel(const float* __restrict__ in, _Float16* __restrict__ out,
                                 int K, int Nc) {
  int id = blockIdx.x * 256 + threadIdx.x;
  if (id < K * Nc) {
    int n = id / K, k = id - n * K;
    out[id] = (_Float16)in[k * Nc + n];
  }
}

// -------------------------------------------------- time / condition embedding
__global__ __launch_bounds__(256) void time_embed_kernel(
    const int* __restrict__ ts, const float* __restrict__ cond,
    const float* __restrict__ te_w1, const float* __restrict__ te_b1,
    const float* __restrict__ te_w2, const float* __restrict__ te_b2,
    const float* __restrict__ ce_w1, const float* __restrict__ ce_b1,
    const float* __restrict__ ce_w2, const float* __restrict__ ce_b2,
    float* __restrict__ temb) {
  __shared__ float xv[TDD], hid[HH], hid2[HH];
  const int j = threadIdx.x;
  for (int b = 0; b < BBATCH; ++b) {
    if (j < TDD) {
      float tval = (float)ts[b];
      int i = j & 63;
      float fr = expf(-logf(10000.0f) / 63.0f * (float)i);
      float a = tval * fr;
      xv[j] = (j < 64) ? sinf(a) : cosf(a);
    }
    __syncthreads();
    float s = te_b1[j];
    for (int k = 0; k < TDD; ++k) s += xv[k] * te_w1[k * HH + j];
    hid[j] = silu_f(s);
    float s2 = ce_b1[j];
    for (int k = 0; k < 3; ++k) s2 += cond[b * 3 + k] * ce_w1[k * HH + j];
    hid2[j] = silu_f(s2);
    __syncthreads();
    if (j < TDD) {
      float o = te_b2[j] + ce_b2[j];
      for (int k = 0; k < HH; ++k) o += hid[k] * te_w2[k * TDD + j] + hid2[k] * ce_w2[k * TDD + j];
      temb[b * TDD + j] = o;
    }
    __syncthreads();
  }
}

__global__ __launch_bounds__(256) void time_proj_kernel(
    const float* __restrict__ temb,
    const float* __restrict__ w1, const float* __restrict__ b1,
    const float* __restrict__ w2, const float* __restrict__ b2,
    float* __restrict__ tp) {
  __shared__ float hid[HH];
  const int j = threadIdx.x;
  for (int bb = 0; bb < BBATCH; ++bb) {
    float s = b1[j];
    for (int k = 0; k < TDD; ++k) s += temb[bb * TDD + k] * w1[k * HH + j];
    hid[j] = silu_f(s);
    __syncthreads();
    float o = b2[j];
    for (int k = 0; k < HH; ++k) o += hid[k] * w2[k * HH + j];
    tp[bb * HH + j] = o;
    __syncthreads();
  }
}

// ------------------------------------------------------------ edge MLP (WMMA)
// 64 edges per workgroup, 8 waves. GEMM1: [64x512]x[512x256] (A gathered from
// h_half via col/row), fused dist/bias/SiLU, GEMM2: [64x256]x[256x256],
// f32 atomic scatter into agg[col].
__global__ __launch_bounds__(256) void edge_mlp_kernel(
    const _Float16* __restrict__ W1T,   // [256][512]  = em_w1[0:512,:]^T f16
    const _Float16* __restrict__ W2T,   // [256][256]  = em_w2^T f16
    const float* __restrict__ w1full,   // em_w1 f32 [513][256] (dist row 512)
    const float* __restrict__ b1,
    const float* __restrict__ b2,
    const _Float16* __restrict__ hh,    // [NN][256] f16
    const int* __restrict__ eidx,       // [2][EE]
    const float* __restrict__ distv,    // [EE]
    float* __restrict__ agg) {          // [NN][256]
  __shared__ __align__(16) _Float16 Abuf[64 * 32];
  __shared__ __align__(16) _Float16 Bbuf[256 * 32];
  __shared__ __align__(16) _Float16 Hid[64 * 256];

  const int t = threadIdx.x;
  const int lane = t & 31;
  const int w = t >> 5;
  const int e0 = blockIdx.x * 64;
  const int mtile = w >> 1;            // 0..3 (16-edge tile)
  const int nbase = (w & 1) * 128;     // 8 N-tiles of 16
  const int lhalf = (lane < 16) ? 0 : 8;
  const int l16 = lane & 15;
  const int bks = lhalf * 2;           // 0 or 16

  const int* __restrict__ rowi = eidx;       // source (x_j)
  const int* __restrict__ coli = eidx + EE;  // target (x_i)

  const int ea = t >> 2;               // edge-local row for A gather
  const int ka = (t & 3) * 8;          // 8-half chunk within 32-K tile
  const int nodeC = coli[e0 + ea];
  const int nodeR = rowi[e0 + ea];

  const v8f vzero = {0.f,0.f,0.f,0.f,0.f,0.f,0.f,0.f};
  v8f acc[8];
#pragma unroll
  for (int i = 0; i < 8; ++i) acc[i] = vzero;

  // ---- GEMM1, K = 512 (first 256 = h[col], next 256 = h[row]) ----
  for (int kb = 0; kb < 512; kb += 32) {
    const int node = (kb < 256) ? nodeC : nodeR;
    const int ko = kb & 255;
    *(h8*)&Abuf[ea * 32 + ka] = *(const h8*)&hh[node * HH + ko + ka];
#pragma unroll
    for (int c = 0; c < 4; ++c)
      *(h8*)&Bbuf[t * 32 + c * 8] = *(const h8*)&W1T[t * 512 + kb + c * 8];
    if (kb + 32 < 512) __builtin_prefetch(&W1T[t * 512 + kb + 32], 0, 1);
    __syncthreads();

    const int arow = mtile * 16 + l16;
    h8 alo = *(const h8*)&Abuf[arow * 32 + lhalf];
    h8 ahi = *(const h8*)&Abuf[arow * 32 + 16 + lhalf];
    v16h af = cat16(alo, ahi);
#pragma unroll
    for (int tt = 0; tt < 8; ++tt) {
      const int ncol = nbase + tt * 16 + l16;
      h8 blo = *(const h8*)&Bbuf[ncol * 32 + lhalf];
      h8 bhi = *(const h8*)&Bbuf[ncol * 32 + 16 + lhalf];
      acc[tt] = wmma_f16(af, cat16(blo, bhi), acc[tt]);
    }
    __syncthreads();
  }

  // ---- epilogue 1: + dist*w1[512] + b1, SiLU -> Hid (f16) ----
#pragma unroll
  for (int tt = 0; tt < 8; ++tt) {
    const int ncol = nbase + tt * 16 + l16;
    const float b1v = b1[ncol];
    const float wd = w1full[512 * HH + ncol];
#pragma unroll
    for (int r = 0; r < 8; ++r) {
      const int m = mtile * 16 + r + lhalf;
      const float d = distv[e0 + m];
      float x = acc[tt][r] + b1v + d * wd;
      Hid[m * HH + ncol] = (_Float16)silu_f(x);
    }
  }
#pragma unroll
  for (int i = 0; i < 8; ++i) acc[i] = vzero;

  // ---- GEMM2, K = 256 ----
  for (int kb = 0; kb < 256; kb += 32) {
#pragma unroll
    for (int c = 0; c < 4; ++c)
      *(h8*)&Bbuf[t * 32 + c * 8] = *(const h8*)&W2T[t * HH + kb + c * 8];
    if (kb + 32 < 256) __builtin_prefetch(&W2T[t * HH + kb + 32], 0, 1);
    __syncthreads();

    const int arow = mtile * 16 + l16;
    h8 alo = *(const h8*)&Hid[arow * HH + kb + lhalf];
    h8 ahi = *(const h8*)&Hid[arow * HH + kb + 16 + lhalf];
    v16h af = cat16(alo, ahi);
#pragma unroll
    for (int tt = 0; tt < 8; ++tt) {
      const int ncol = nbase + tt * 16 + l16;
      h8 blo = *(const h8*)&Bbuf[ncol * 32 + lhalf];
      h8 bhi = *(const h8*)&Bbuf[ncol * 32 + 16 + lhalf];
      acc[tt] = wmma_f16(af, cat16(blo, bhi), acc[tt]);
    }
    __syncthreads();
  }

  // ---- epilogue 2: + b2, atomic scatter (sum; divide by counts later) ----
  int nd[8];
#pragma unroll
  for (int r = 0; r < 8; ++r) nd[r] = coli[e0 + mtile * 16 + r + lhalf];
#pragma unroll
  for (int tt = 0; tt < 8; ++tt) {
    const int ncol = nbase + tt * 16 + l16;
    const float b2v = b2[ncol];
#pragma unroll
    for (int r = 0; r < 8; ++r)
      atomicAdd(&agg[nd[r] * HH + ncol], acc[tt][r] + b2v);
  }
}

// --------------------------------------------------- node update (WMMA + LN)
// 32 nodes per workgroup. h_new = node_mlp(h) + agg/counts; h = LN(h + h_new).
__global__ __launch_bounds__(256) void node_update_kernel(
    const _Float16* __restrict__ W1T,   // nm_w1^T f16 [256][256]
    const _Float16* __restrict__ W2T,   // nm_w2^T f16 [256][256]
    const float* __restrict__ b1,
    const float* __restrict__ b2,
    float* __restrict__ h,
    _Float16* __restrict__ hh,
    float* __restrict__ agg,            // read then zeroed for next layer
    const float* __restrict__ icnt,
    const float* __restrict__ lng,
    const float* __restrict__ lnb) {
  __shared__ __align__(16) char smem[37120];
  _Float16* Abuf = (_Float16*)(smem);           // [32*32]   2048 B
  _Float16* Bbuf = (_Float16*)(smem + 2048);    // [256*32] 16384 B
  _Float16* Hid  = (_Float16*)(smem + 18432);   // [32*256] 16384 B
  float* xbuf = (float*)(smem);                 // [32*256] 32768 B (overlay)
  float* red  = (float*)(smem + 34816);         // [512]
  float* mv   = (float*)(smem + 36864);         // [64]

  const int t = threadIdx.x;
  const int lane = t & 31;
  const int w = t >> 5;
  const int n0 = blockIdx.x * 32;
  const int mtile = w >> 2;            // 0..1
  const int nbase = (w & 3) * 64;      // 4 N-tiles of 16
  const int lhalf = (lane < 16) ? 0 : 8;
  const int l16 = lane & 15;
  const int bks = lhalf * 2;

  const int ar = t >> 2;               // 0..63 (only <32 used)
  const int ak = (t & 3) * 8;
  int anode = n0 + ar; if (anode >= NN) anode = NN - 1;

  const v8f vzero = {0.f,0.f,0.f,0.f,0.f,0.f,0.f,0.f};
  v8f acc[4];
#pragma unroll
  for (int i = 0; i < 4; ++i) acc[i] = vzero;

  // ---- GEMM1: hidden = silu(h @ nm_w1 + b1) ----
  for (int kb = 0; kb < 256; kb += 32) {
    if (t < 128)
      *(h8*)&Abuf[ar * 32 + ak] = *(const h8*)&hh[anode * HH + kb + ak];
#pragma unroll
    for (int c = 0; c < 4; ++c)
      *(h8*)&Bbuf[t * 32 + c * 8] = *(const h8*)&W1T[t * HH + kb + c * 8];
    __syncthreads();

    const int arow = mtile * 16 + l16;
    h8 alo = *(const h8*)&Abuf[arow * 32 + lhalf];
    h8 ahi = *(const h8*)&Abuf[arow * 32 + 16 + lhalf];
    v16h af = cat16(alo, ahi);
#pragma unroll
    for (int tt = 0; tt < 4; ++tt) {
      const int ncol = nbase + tt * 16 + l16;
      h8 blo = *(const h8*)&Bbuf[ncol * 32 + lhalf];
      h8 bhi = *(const h8*)&Bbuf[ncol * 32 + 16 + lhalf];
      acc[tt] = wmma_f16(af, cat16(blo, bhi), acc[tt]);
    }
    __syncthreads();
  }
#pragma unroll
  for (int tt = 0; tt < 4; ++tt) {
    const int ncol = nbase + tt * 16 + l16;
    const float b1v = b1[ncol];
#pragma unroll
    for (int r = 0; r < 8; ++r) {
      const int m = mtile * 16 + r + lhalf;
      Hid[m * HH + ncol] = (_Float16)silu_f(acc[tt][r] + b1v);
    }
  }
#pragma unroll
  for (int i = 0; i < 4; ++i) acc[i] = vzero;

  // ---- GEMM2: hidden @ nm_w2 ----
  for (int kb = 0; kb < 256; kb += 32) {
#pragma unroll
    for (int c = 0; c < 4; ++c)
      *(h8*)&Bbuf[t * 32 + c * 8] = *(const h8*)&W2T[t * HH + kb + c * 8];
    __syncthreads();

    const int arow = mtile * 16 + l16;
    h8 alo = *(const h8*)&Hid[arow * HH + kb + lhalf];
    h8 ahi = *(const h8*)&Hid[arow * HH + kb + 16 + lhalf];
    v16h af = cat16(alo, ahi);
#pragma unroll
    for (int tt = 0; tt < 4; ++tt) {
      const int ncol = nbase + tt * 16 + l16;
      h8 blo = *(const h8*)&Bbuf[ncol * 32 + lhalf];
      h8 bhi = *(const h8*)&Bbuf[ncol * 32 + 16 + lhalf];
      acc[tt] = wmma_f16(af, cat16(blo, bhi), acc[tt]);
    }
    __syncthreads();
  }

  // ---- residual + mean-agg; zero agg; stage x for LN ----
#pragma unroll
  for (int tt = 0; tt < 4; ++tt) {
    const int ncol = nbase + tt * 16 + l16;
    const float b2v = b2[ncol];
#pragma unroll
    for (int r = 0; r < 8; ++r) {
      const int m = mtile * 16 + r + lhalf;
      const int node = n0 + m;
      if (node < NN) {
        const int idx = node * HH + ncol;
        float x = h[idx] + acc[tt][r] + b2v + agg[idx] * icnt[node];
        agg[idx] = 0.0f;
        xbuf[m * HH + ncol] = x;
      }
    }
  }
  __syncthreads();

  // ---- LayerNorm (8 threads per row) ----
  const int rrow = t >> 3, seg = t & 7;
  {
    float sm = 0.f, sq = 0.f;
#pragma unroll
    for (int jj = 0; jj < 32; ++jj) {
      float v = xbuf[rrow * HH + seg * 32 + jj];
      sm += v; sq += v * v;
    }
    red[rrow * 8 + seg] = sm;
    red[256 + rrow * 8 + seg] = sq;
  }
  __syncthreads();
  if (seg == 0) {
    float s = 0.f, q = 0.f;
#pragma unroll
    for (int i = 0; i < 8; ++i) { s += red[rrow * 8 + i]; q += red[256 + rrow * 8 + i]; }
    float mean = s * (1.0f / HH);
    float var = q * (1.0f / HH) - mean * mean;
    mv[rrow * 2] = mean;
    mv[rrow * 2 + 1] = rsqrtf(var + 1e-5f);
  }
  __syncthreads();
  const int node = n0 + rrow;
  if (node < NN) {
    const float mean = mv[rrow * 2], inv = mv[rrow * 2 + 1];
    for (int jj = 0; jj < 32; ++jj) {
      const int j = seg * 32 + jj;
      float y = (xbuf[rrow * HH + j] - mean) * inv * lng[j] + lnb[j];
      h[node * HH + j] = y;
      hh[node * HH + j] = (_Float16)y;
    }
  }
}

// ----------------------------------------------------------------- out heads
__global__ __launch_bounds__(256) void head_kernel(
    const float* __restrict__ h,
    const float* __restrict__ ao_w1, const float* __restrict__ ao_b1,
    const float* __restrict__ ao_w2, const float* __restrict__ ao_b2,
    const float* __restrict__ co_w1, const float* __restrict__ co_b1,
    const float* __restrict__ co_w2, const float* __restrict__ co_b2,
    float* __restrict__ out_atom, float* __restrict__ out_coord) {
  __shared__ float hv[HH], ha[HH], hc[HH];
  const int n = blockIdx.x, j = threadIdx.x;
  hv[j] = h[(size_t)n * HH + j];
  __syncthreads();
  float s = ao_b1[j], s2 = co_b1[j];
  for (int k = 0; k < HH; ++k) {
    float v = hv[k];
    s  += v * ao_w1[k * HH + j];
    s2 += v * co_w1[k * HH + j];
  }
  ha[j] = silu_f(s);
  hc[j] = silu_f(s2);
  __syncthreads();
  if (j < NAT_) {
    float o = ao_b2[j];
    for (int k = 0; k < HH; ++k) o += ha[k] * ao_w2[k * NAT_ + j];
    out_atom[(size_t)n * NAT_ + j] = o;
  }
  if (j >= 128 && j < 131) {
    int c = j - 128;
    float o = co_b2[c];
    for (int k = 0; k < HH; ++k) o += hc[k] * co_w2[k * 3 + c];
    out_coord[(size_t)n * 3 + c] = o;
  }
}

// -------------------------------------------------------------------- driver
extern "C" void kernel_launch(void* const* d_in, const int* in_sizes, int n_in,
                              void* d_out, int out_size, void* d_ws, size_t ws_size,
                              hipStream_t stream) {
  (void)in_sizes; (void)n_in; (void)out_size; (void)ws_size;
  const int*   x_atoms    = (const int*)  d_in[0];
  const float* pos        = (const float*)d_in[1];
  const int*   edge_index = (const int*)  d_in[2];
  const int*   batch_idx  = (const int*)  d_in[3];
  const int*   timesteps  = (const int*)  d_in[4];
  const float* conditions = (const float*)d_in[5];
  const float* emb   = (const float*)d_in[6];
  const float* te_w1 = (const float*)d_in[7];
  const float* te_b1 = (const float*)d_in[8];
  const float* te_w2 = (const float*)d_in[9];
  const float* te_b2 = (const float*)d_in[10];
  const float* ce_w1 = (const float*)d_in[11];
  const float* ce_b1 = (const float*)d_in[12];
  const float* ce_w2 = (const float*)d_in[13];
  const float* ce_b2 = (const float*)d_in[14];
  const float* tp_w1 = (const float*)d_in[15];
  const float* tp_b1 = (const float*)d_in[16];
  const float* tp_w2 = (const float*)d_in[17];
  const float* tp_b2 = (const float*)d_in[18];
  const float* nm_w1 = (const float*)d_in[19];
  const float* nm_b1 = (const float*)d_in[20];
  const float* nm_w2 = (const float*)d_in[21];
  const float* nm_b2 = (const float*)d_in[22];
  const float* em_w1 = (const float*)d_in[23];
  const float* em_b1 = (const float*)d_in[24];
  const float* em_w2 = (const float*)d_in[25];
  const float* em_b2 = (const float*)d_in[26];
  const float* ln_g  = (const float*)d_in[27];
  const float* ln_b  = (const float*)d_in[28];
  const float* ao_w1 = (const float*)d_in[29];
  const float* ao_b1 = (const float*)d_in[30];
  const float* ao_w2 = (const float*)d_in[31];
  const float* ao_b2 = (const float*)d_in[32];
  const float* co_w1 = (const float*)d_in[33];
  const float* co_b1 = (const float*)d_in[34];
  const float* co_w2 = (const float*)d_in[35];
  const float* co_b2 = (const float*)d_in[36];

  char* ws = (char*)d_ws;
  size_t off = 0;
  auto carve = [&](size_t bytes) {
    char* p = ws + off;
    off += (bytes + 255) & ~(size_t)255;
    return p;
  };
  float*    wH    = (float*)   carve((size_t)NN * HH * 4);
  _Float16* wHH   = (_Float16*)carve((size_t)NN * HH * 2);
  float*    wAGG  = (float*)   carve((size_t)NN * HH * 4);
  float*    wDIST = (float*)   carve((size_t)EE * 4);
  float*    wCNT  = (float*)   carve((size_t)NN * 4);
  float*    wTEMB = (float*)   carve((size_t)BBATCH * TDD * 4);
  float*    wTP   = (float*)   carve((size_t)BBATCH * HH * 4);
  _Float16* wW1T  = (_Float16*)carve((size_t)HH * 512 * 2);
  _Float16* wW2T  = (_Float16*)carve((size_t)HH * HH * 2);
  _Float16* wNW1T = (_Float16*)carve((size_t)HH * HH * 2);
  _Float16* wNW2T = (_Float16*)carve((size_t)HH * HH * 2);

  zero_f32_kernel<<<(NN * HH) / 256, 256, 0, stream>>>(wAGG, NN * HH);
  zero_f32_kernel<<<(NN + 255) / 256, 256, 0, stream>>>(wCNT, NN);
  embed_kernel<<<(NN * HH) / 256, 256, 0, stream>>>(x_atoms, emb, wH, wHH);
  dist_kernel<<<(EE + 255) / 256, 256, 0, stream>>>(pos, edge_index, wDIST);
  count_kernel<<<(EE + 255) / 256, 256, 0, stream>>>(edge_index, wCNT);
  invcnt_kernel<<<(NN + 255) / 256, 256, 0, stream>>>(wCNT, NN);
  time_embed_kernel<<<1, 256, 0, stream>>>(timesteps, conditions,
      te_w1, te_b1, te_w2, te_b2, ce_w1, ce_b1, ce_w2, ce_b2, wTEMB);

  for (int b = 0; b < NBLK; ++b) {
    time_proj_kernel<<<1, 256, 0, stream>>>(wTEMB,
        tp_w1 + (size_t)b * TDD * HH, tp_b1 + b * HH,
        tp_w2 + (size_t)b * HH * HH, tp_b2 + b * HH, wTP);
    addtp_kernel<<<(NN * HH) / 256, 256, 0, stream>>>(wH, wHH, wTP, batch_idx);
    for (int l = 0; l < NLAY; ++l) {
      const int bl = b * NLAY + l;
      const float* ew1 = em_w1 + (size_t)bl * (2 * HH + 1) * HH;
      transpose_kernel<<<(512 * HH) / 256, 256, 0, stream>>>(ew1, wW1T, 512, HH);
      transpose_kernel<<<(HH * HH) / 256, 256, 0, stream>>>(
          em_w2 + (size_t)bl * HH * HH, wW2T, HH, HH);
      transpose_kernel<<<(HH * HH) / 256, 256, 0, stream>>>(
          nm_w1 + (size_t)bl * HH * HH, wNW1T, HH, HH);
      transpose_kernel<<<(HH * HH) / 256, 256, 0, stream>>>(
          nm_w2 + (size_t)bl * HH * HH, wNW2T, HH, HH);
      edge_mlp_kernel<<<EE / 64, 256, 0, stream>>>(wW1T, wW2T, ew1,
          em_b1 + bl * HH, em_b2 + bl * HH, wHH, edge_index, wDIST, wAGG);
      node_update_kernel<<<(NN + 31) / 32, 256, 0, stream>>>(wNW1T, wNW2T,
          nm_b1 + bl * HH, nm_b2 + bl * HH, wH, wHH, wAGG, wCNT,
          ln_g + bl * HH, ln_b + bl * HH);
    }
  }
  head_kernel<<<NN, 256, 0, stream>>>(wH, ao_w1, ao_b1, ao_w2, ao_b2,
      co_w1, co_b1, co_w2, co_b2,
      (float*)d_out, (float*)d_out + (size_t)NN * NAT_);
}
